// DyGraphConv2d_8031588843842
// MI455X (gfx1250) — matrix operand order repaired and checked by reading, not compiled, not run on hardware
//
#include <hip/hip_runtime.h>
#include <math.h>

#define BB   32
#define CC   192
#define NN   784
#define KNN  9
#define COUT 384
#define CG   96
#define GR   4

typedef __attribute__((ext_vector_type(16))) _Float16 v16h;
typedef __attribute__((ext_vector_type(8)))  float    v8f;

__device__ __forceinline__ void top9_insert(unsigned (&top)[9], unsigned key) {
#pragma unroll
  for (int j = 0; j < 9; j++) {
    unsigned mx = top[j] >= key ? top[j] : key;
    unsigned mn = top[j] >= key ? key : top[j];
    top[j] = mx; key = mn;
  }
}

__device__ __forceinline__ void scan_one(unsigned (&top)[9], float s, unsigned n) {
  unsigned u = __float_as_uint(s);
  u = (u & 0x80000000u) ? ~u : (u | 0x80000000u);     // order-preserving float->uint
  unsigned key = (u & 0xFFFFFC00u) | (1023u - n);     // score hi 22b, ~index lo 10b
  if (key > top[8]) top9_insert(top, key);            // fast reject; ties -> smaller n
}

// ---------------- weight pack f32 -> f16 ----------------
__global__ void k_pack_w(const float* __restrict__ w, _Float16* __restrict__ wh, int n) {
  int i = blockIdx.x * 256 + threadIdx.x;
  if (i < n) wh[i] = (_Float16)w[i];
}

// ---------------- L2-normalize, store [b][n][c] f16 ----------------
__global__ void k_normalize(const float* __restrict__ x, _Float16* __restrict__ xn) {
  int t = blockIdx.x * blockDim.x + threadIdx.x;
  if (t >= BB * NN) return;
  int b = t / NN, n = t - b * NN;
  const float* xb = x + (size_t)b * CC * NN + n;
  float ss = 0.f;
#pragma unroll 4
  for (int c = 0; c < CC; c++) { float v = xb[(size_t)c * NN]; ss = fmaf(v, v, ss); }
  float s = 1.0f / fmaxf(sqrtf(ss), 1e-12f);
  _Float16* dst = xn + ((size_t)b * NN + n) * CC;
#pragma unroll 4
  for (int c = 0; c < CC; c++) dst[c] = (_Float16)(xb[(size_t)c * NN] * s);
}

// ---------------- KNN: WMMA inner products + fused top-9 (u32 keys) ----------------
__global__ __launch_bounds__(32) void k_knn(const _Float16* __restrict__ xn, int* __restrict__ nnidx) {
  __shared__ float tile[16][20];          // stride 20: 16B-aligned rows, conflict-free stores
  __shared__ unsigned mrg[32][9];
  int b  = blockIdx.x;
  int mt = blockIdx.y;
  int l = threadIdx.x;
  int half = l >> 4, lm = l & 15;
  const _Float16* xb = xn + (size_t)b * NN * CC;

  // A fragments: 16 rows x 192 K, resident in registers (6 k-steps of 32)
  v16h A[6];
  {
    int m = mt * 16 + lm;
#pragma unroll
    for (int ks = 0; ks < 6; ks++) {
      const _Float16* p = xb + (size_t)m * CC + ks * 32 + half * 8;
#pragma unroll
      for (int e = 0; e < 8; e++) { A[ks][e] = p[e]; A[ks][8 + e] = p[16 + e]; }
    }
  }
  unsigned top[9];
#pragma unroll
  for (int j = 0; j < 9; j++) top[j] = 0u;

  // double-buffered B fragments: prefetch tile nt+1 while selecting tile nt
  v16h Bcur[6], Bnext[6];
  {
    const _Float16* pb = xb + (size_t)lm * CC + 8 * half;
#pragma unroll
    for (int ks = 0; ks < 6; ks++)
#pragma unroll
      for (int e = 0; e < 16; e++) Bcur[ks][e] = pb[ks * 32 + e + (e >= 8 ? 8 : 0)];
  }

  for (int nt = 0; nt < 49; nt++) {
    v8f acc = {};
#pragma unroll
    for (int ks = 0; ks < 6; ks++)
      acc = __builtin_amdgcn_wmma_f32_16x16x32_f16(false, A[ks], false, Bcur[ks],
                                                   (short)0, acc, false, false);
    if (nt < 48) {
      const _Float16* pb = xb + (size_t)((nt + 1) * 16 + lm) * CC + 8 * half;
#pragma unroll
      for (int ks = 0; ks < 6; ks++)
#pragma unroll
        for (int e = 0; e < 16; e++) Bnext[ks][e] = pb[ks * 32 + e + (e >= 8 ? 8 : 0)];
    }
    // transpose 16x16 tile through LDS so each lane scans half a row
#pragma unroll
    for (int v = 0; v < 8; v++) tile[v + 8 * half][lm] = acc[v];
    __syncthreads();
    // vectorized row-slice read: two b128 LDS loads per lane
    float4 r0 = *(const float4*)&tile[lm][half * 8];
    float4 r1 = *(const float4*)&tile[lm][half * 8 + 4];
    unsigned n0 = (unsigned)(nt * 16 + half * 8);
    scan_one(top, r0.x, n0 + 0);
    scan_one(top, r0.y, n0 + 1);
    scan_one(top, r0.z, n0 + 2);
    scan_one(top, r0.w, n0 + 3);
    scan_one(top, r1.x, n0 + 4);
    scan_one(top, r1.y, n0 + 5);
    scan_one(top, r1.z, n0 + 6);
    scan_one(top, r1.w, n0 + 7);
    __syncthreads();
#pragma unroll
    for (int ks = 0; ks < 6; ks++) Bcur[ks] = Bnext[ks];
  }
  // merge the two half-row scanners (lane l and l+16 share row l&15)
#pragma unroll
  for (int j = 0; j < 9; j++) mrg[l][j] = top[j];
  __syncthreads();
  if (half == 0) {
#pragma unroll
    for (int j = 0; j < 9; j++) {
      unsigned key = mrg[l + 16][j];
      if (key > top[8]) top9_insert(top, key);
    }
    int row = mt * 16 + lm;
    int* dst = nnidx + ((size_t)b * NN + row) * KNN;
#pragma unroll
    for (int j = 0; j < 9; j++) dst[j] = (int)(1023u - (top[j] & 1023u));
  }
}

// ------- max-relative gather + pack interleaved h=[xf,rel] as f16 [b][n][2C] -------
__global__ void k_relpack(const float* __restrict__ x, const int* __restrict__ nnidx,
                          _Float16* __restrict__ h16) {
  int t = blockIdx.x * blockDim.x + threadIdx.x;
  if (t >= BB * NN) return;
  int b = t / NN, n = t - b * NN;
  int idx[KNN];
  const int* ip = nnidx + ((size_t)b * NN + n) * KNN;
#pragma unroll
  for (int k = 0; k < KNN; k++) idx[k] = ip[k];
  const float* xb = x + (size_t)b * CC * NN;
  _Float16* hp = h16 + (size_t)t * (2 * CC);
  for (int c = 0; c < CC; c++) {
    const float* row = xb + (size_t)c * NN;
    float xi = row[n];
    float mv = row[idx[0]];
#pragma unroll
    for (int k = 1; k < KNN; k++) mv = fmaxf(mv, row[idx[k]]);
    hp[2 * c]     = (_Float16)xi;          // h[2c]   = xf[c]
    hp[2 * c + 1] = (_Float16)(mv - xi);   // h[2c+1] = rel[c]
  }
}

// ---------------- grouped 1x1 conv via WMMA, writes pre-BN to d_out ----------------
__global__ __launch_bounds__(32) void k_conv(const _Float16* __restrict__ h16,
                                             const _Float16* __restrict__ wh,
                                             const float* __restrict__ bias,
                                             float* __restrict__ out) {
  int l = threadIdx.x, half = l >> 4, lm = l & 15;
  int nch = blockIdx.x;               // 7 chunks of 7 n-tiles
  int mt  = blockIdx.y;               // 6 m-tiles of 16
  int bg  = blockIdx.z;               // b*4 + g
  int b = bg >> 2, g = bg & 3;
  int co0 = g * CG + mt * 16;

  v16h A[3];
  {
    int rowc = co0 + lm;
#pragma unroll
    for (int ks = 0; ks < 3; ks++) {
      const _Float16* p = wh + (size_t)rowc * CG + ks * 32 + half * 8;
#pragma unroll
      for (int e = 0; e < 8; e++) { A[ks][e] = p[e]; A[ks][8 + e] = p[16 + e]; }
    }
  }
  float biasv[8];
#pragma unroll
  for (int v = 0; v < 8; v++) biasv[v] = bias[co0 + v + 8 * half];

  for (int t = 0; t < 7; t++) {
    int n = (nch * 7 + t) * 16 + lm;
    const _Float16* ph = h16 + ((size_t)b * NN + n) * (2 * CC) + g * CG + 8 * half;
    v16h Bf[3];
#pragma unroll
    for (int ks = 0; ks < 3; ks++)
#pragma unroll
      for (int e = 0; e < 16; e++) Bf[ks][e] = ph[ks * 32 + e + (e >= 8 ? 8 : 0)];
    v8f acc = {};
#pragma unroll
    for (int ks = 0; ks < 3; ks++)
      acc = __builtin_amdgcn_wmma_f32_16x16x32_f16(false, A[ks], false, Bf[ks],
                                                   (short)0, acc, false, false);
#pragma unroll
    for (int v = 0; v < 8; v++) {
      out[((size_t)b * COUT + co0 + v + 8 * half) * NN + n] = acc[v] + biasv[v];
    }
  }
}

// ---------------- per-channel batch stats ----------------
__global__ __launch_bounds__(256) void k_stats(const float* __restrict__ out, float* __restrict__ stats) {
  int co = blockIdx.x, t = threadIdx.x;
  float s = 0.f, s2 = 0.f;
  for (int i = t; i < BB * NN; i += 256) {
    int b = i / NN, n = i - b * NN;
    float v = out[((size_t)b * COUT + co) * NN + n];
    s += v; s2 = fmaf(v, v, s2);
  }
  __shared__ float sh[256], sh2[256];
  sh[t] = s; sh2[t] = s2;
  __syncthreads();
  for (int w = 128; w > 0; w >>= 1) {
    if (t < w) { sh[t] += sh[t + w]; sh2[t] += sh2[t + w]; }
    __syncthreads();
  }
  if (t == 0) {
    float inv = 1.0f / (float)(BB * NN);
    float mean = sh[0] * inv;
    float var  = sh2[0] * inv - mean * mean;
    stats[co] = mean;
    stats[COUT + co] = var;
  }
}

// ---------------- BN + exact GELU, in place ----------------
__global__ void k_bngelu(float* __restrict__ out, const float* __restrict__ stats,
                         const float* __restrict__ gamma, const float* __restrict__ beta, int total) {
  int i = blockIdx.x * 256 + threadIdx.x;
  if (i >= total) return;
  int co = (i / NN) % COUT;
  float v = out[i];
  v = (v - stats[co]) * rsqrtf(stats[COUT + co] + 1e-5f);
  v = v * gamma[co] + beta[co];
  out[i] = 0.5f * v * (1.0f + erff(v * 0.70710678118654752440f));
}

extern "C" void kernel_launch(void* const* d_in, const int* in_sizes, int n_in,
                              void* d_out, int out_size, void* d_ws, size_t ws_size,
                              hipStream_t stream) {
  const float* x      = (const float*)d_in[0];
  const float* conv_w = (const float*)d_in[1];
  const float* conv_b = (const float*)d_in[2];
  const float* gamma  = (const float*)d_in[3];
  const float* beta   = (const float*)d_in[4];
  float* out = (float*)d_out;
  char* ws = (char*)d_ws;

  size_t off = 0;
  auto take = [&](size_t bytes) -> char* {
    char* p = ws + off;
    off = (off + bytes + 255) & ~(size_t)255;
    return p;
  };
  _Float16* xn    = (_Float16*)take((size_t)BB * NN * CC * 2);        // ~9.6 MB
  _Float16* wh    = (_Float16*)take((size_t)COUT * CG * 2);           // 72 KB
  int*      nnbuf = (int*)     take((size_t)BB * NN * KNN * 4);       // ~0.9 MB
  _Float16* h16   = (_Float16*)take((size_t)BB * NN * 2 * CC * 2);    // ~19.3 MB
  float*    stats = (float*)   take((size_t)2 * COUT * 4);            // 3 KB
  (void)ws_size; (void)in_sizes; (void)n_in; (void)out_size;

  k_pack_w   <<<(COUT * CG + 255) / 256, 256, 0, stream>>>(conv_w, wh, COUT * CG);
  k_normalize<<<(BB * NN + 255) / 256, 256, 0, stream>>>(x, xn);
  k_knn      <<<dim3(BB, NN / 16), 32, 0, stream>>>(xn, nnbuf);
  k_relpack  <<<(BB * NN + 255) / 256, 256, 0, stream>>>(x, nnbuf, h16);
  k_conv     <<<dim3(7, 6, BB * GR), 32, 0, stream>>>(h16, wh, conv_b, out);
  k_stats    <<<COUT, 256, 0, stream>>>(out, stats);
  int total = BB * COUT * NN;
  k_bngelu   <<<(total + 255) / 256, 256, 0, stream>>>(out, stats, gamma, beta, total);
}
